// MultiHeadAttention_35845797052879
// MI455X (gfx1250) — matrix operand order repaired and checked by reading, not compiled, not run on hardware
//
#include <hip/hip_runtime.h>
#include <hip/hip_fp16.h>
#include <stdint.h>

#define B_  4
#define S_  2048
#define D_  1024
#define H_  16
#define HD_ 64

typedef _Float16 f16;
typedef _Float16 half16 __attribute__((ext_vector_type(16)));
typedef float    float8 __attribute__((ext_vector_type(8)));
typedef int      vi4    __attribute__((vector_size(16)));

// ---- CDNA5 async global->LDS copy (GLOBAL_LOAD_ASYNC_TO_LDS_B128) ----------
#if defined(__has_builtin)
#if __has_builtin(__builtin_amdgcn_global_load_async_to_lds_b128)
#define HAVE_ASYNC_LDS 1
#endif
#endif
#ifndef HAVE_ASYNC_LDS
#define HAVE_ASYNC_LDS 0
#endif

#if HAVE_ASYNC_LDS
#if __has_builtin(__builtin_amdgcn_s_wait_asynccnt)
#define WAIT_ASYNC(n) __builtin_amdgcn_s_wait_asynccnt(n)
#else
#define WAIT_ASYNC(n) asm volatile("s_wait_asynccnt %0" ::"i"(n) : "memory")
#endif
#else
#define WAIT_ASYNC(n) ((void)0)
#endif

static __device__ __forceinline__ void cp16(f16* l, const f16* g) {
#if HAVE_ASYNC_LDS
  __builtin_amdgcn_global_load_async_to_lds_b128(
      (__attribute__((address_space(1))) vi4*)(uintptr_t)g,
      (__attribute__((address_space(3))) vi4*)(uintptr_t)l, 0, 0);
#else
  *(float4*)l = *(const float4*)g;
#endif
}

static __device__ __forceinline__ float8 zero8() {
  float8 z = {0.f,0.f,0.f,0.f,0.f,0.f,0.f,0.f};
  return z;
}

// A-matrix 16x32 f16 fragment (M x K). Lane L holds row m = L&15; VGPRs 0..3
// hold K = khalf+0..7 pairs, VGPRs 4..7 hold K = khalf+16..23, khalf=8 for
// lanes 16..31 (CDNA5 ISA).
static __device__ __forceinline__ half16 load_a_frag(const f16* base, int strideElem, int lane) {
  int m = lane & 15;
  int khalf = (lane >> 4) ? 8 : 0;
  const f16* row = base + m * strideElem;
  union { half16 h; uint32_t u[8]; } r;
#pragma unroll
  for (int i = 0; i < 8; ++i) {
    int k = ((i < 4) ? (2 * i) : (16 + 2 * (i - 4))) + khalf;
    r.u[i] = *(const uint32_t*)(row + k);
  }
  return r.h;
}

// B-matrix 32x16 f16 fragment (K x N). Lane L holds column n = L&15; element s
// holds K = khalf+s (s<8) or khalf+16+(s-8) (s>=8), khalf=8 for lanes 16..31.
static __device__ __forceinline__ half16 load_b_frag(const f16* base, int strideElem, int lane) {
  const f16* col = base + (lane & 15);
  int khalf = (lane >> 4) * 8;
  half16 r;
#pragma unroll
  for (int s = 0; s < 16; ++s) r[s] = col[(((s < 8) ? s : (8 + s)) + khalf) * strideElem];
  return r;
}

static __device__ __forceinline__ float8 wmma_f16(half16 a, half16 b, float8 c) {
  return __builtin_amdgcn_wmma_f32_16x16x32_f16(false, a, false, b, (short)0, c, false, false);
}

// ---------------------------------------------------------------- converts
__global__ __launch_bounds__(256) void cvt_f32_to_f16(const float* __restrict__ in,
                                                      f16* __restrict__ out, int n) {
  int i = blockIdx.x * 256 + threadIdx.x;
  if (i < n) out[i] = (f16)in[i];
}

__global__ __launch_bounds__(256) void fuse_wqkv(const float* __restrict__ Wq,
                                                 const float* __restrict__ Wk,
                                                 const float* __restrict__ Wv,
                                                 f16* __restrict__ Wh) {
  int i = blockIdx.x * 256 + threadIdx.x;
  if (i >= D_ * 3 * D_) return;
  int d = i / (3 * D_), c = i % (3 * D_);
  int part = c >> 10, w = c & 1023;
  const float* src = (part == 0) ? Wq : (part == 1) ? Wk : Wv;
  Wh[i] = (f16)src[d * D_ + w];
}

// ---------------------------------------------------------------- GEMM core
// Block: 256 threads (8 waves, 4x2). Tile: 128(M) x 128(N). K-step 32.
// Wave (wm,wn): 32 rows x 64 cols -> 8 WMMAs per k-step. Double-buffered LDS
// fed by async global->LDS copies.
#define LDA_S 40    // 80B stride, 16B aligned
#define LDB_S 136   // 272B stride, 16B aligned

static __device__ __forceinline__ void gemm_stage(
    const f16* __restrict__ A, const f16* __restrict__ Bm,
    int N, int K, int row0, int col0, int k0,
    f16* Alds, f16* Blds, int tid) {
  int c0 = tid, c1 = tid + 256;
  // A tile: 128 rows x 32 cols = 512 8-elt chunks
  { int r = c0 >> 2, col8 = (c0 & 3) * 8;
    cp16(&Alds[r * LDA_S + col8], &A[(size_t)(row0 + r) * K + k0 + col8]); }
  { int r = c1 >> 2, col8 = (c1 & 3) * 8;
    cp16(&Alds[r * LDA_S + col8], &A[(size_t)(row0 + r) * K + k0 + col8]); }
  // B tile: 32 rows x 128 cols = 512 8-elt chunks
  { int r = c0 >> 4, col8 = (c0 & 15) * 8;
    cp16(&Blds[r * LDB_S + col8], &Bm[(size_t)(k0 + r) * N + col0 + col8]); }
  { int r = c1 >> 4, col8 = (c1 & 15) * 8;
    cp16(&Blds[r * LDB_S + col8], &Bm[(size_t)(k0 + r) * N + col0 + col8]); }
}

static __device__ __forceinline__ void gemm_mainloop(
    const f16* __restrict__ A, const f16* __restrict__ Bm,
    int N, int K, int row0, int col0,
    f16* Alds, f16* Blds, float8 acc[2][4], int tid) {
  const int nIter = K / 32;
  int lane = tid & 31, wave = tid >> 5;
  int wm = (wave >> 1) * 32, wn = (wave & 1) * 64;
  gemm_stage(A, Bm, N, K, row0, col0, 0, Alds, Blds, tid);
  for (int kk = 0; kk < nIter; ++kk) {
    f16* Ac = Alds + (kk & 1) * (128 * LDA_S);
    f16* Bc = Blds + (kk & 1) * (32 * LDB_S);
    if (kk + 1 < nIter) {
      f16* An = Alds + ((kk + 1) & 1) * (128 * LDA_S);
      f16* Bn = Blds + ((kk + 1) & 1) * (32 * LDB_S);
      gemm_stage(A, Bm, N, K, row0, col0, (kk + 1) * 32, An, Bn, tid);
      WAIT_ASYNC(4);   // this thread's previous-stage copies complete (in-order)
    } else {
      WAIT_ASYNC(0);
    }
    __syncthreads();
    half16 a0 = load_a_frag(&Ac[wm * LDA_S], LDA_S, lane);
    half16 a1 = load_a_frag(&Ac[(wm + 16) * LDA_S], LDA_S, lane);
#pragma unroll
    for (int j = 0; j < 4; ++j) {
      half16 b = load_b_frag(&Bc[wn + j * 16], LDB_S, lane);
      acc[0][j] = wmma_f16(a0, b, acc[0][j]);
      acc[1][j] = wmma_f16(a1, b, acc[1][j]);
    }
    __syncthreads();
  }
}

// QKV projection: C[8192][3072] = Xh * Wh. Epilogue adds bias, scales Q by
// 0.125, writes Q/V as [b][h][s][e] and K pre-transposed as [b][h][e][s].
__global__ __launch_bounds__(256) void gemm_qkv(
    const f16* __restrict__ Xh, const f16* __restrict__ Wh,
    const float* __restrict__ bq, const float* __restrict__ bk,
    const float* __restrict__ bv,
    f16* __restrict__ Qh, f16* __restrict__ KhT, f16* __restrict__ Vh) {
  __shared__ f16 Alds[2 * 128 * LDA_S];
  __shared__ f16 Blds[2 * 32 * LDB_S];
  int tid = threadIdx.x, lane = tid & 31, wave = tid >> 5;
  int row0 = blockIdx.y * 128, col0 = blockIdx.x * 128;
  float8 acc[2][4] = {{zero8(), zero8(), zero8(), zero8()},
                      {zero8(), zero8(), zero8(), zero8()}};
  gemm_mainloop(Xh, Wh, 3 * D_, D_, row0, col0, Alds, Blds, acc, tid);

  int n = lane & 15, mh = (lane >> 4) * 8;
  int wm = (wave >> 1) * 32, wn = (wave & 1) * 64;
#pragma unroll
  for (int sub = 0; sub < 2; ++sub)
#pragma unroll
    for (int j = 0; j < 4; ++j)
#pragma unroll
      for (int r = 0; r < 8; ++r) {
        int grow = row0 + wm + sub * 16 + r + mh;
        int gcol = col0 + wn + j * 16 + n;
        int part = gcol >> 10, w = gcol & 1023;
        float bias = (part == 0) ? bq[w] : (part == 1) ? bk[w] : bv[w];
        float val = acc[sub][j][r] + bias;
        if (part == 0) val *= 0.125f;
        int h = w >> 6, e = w & 63;
        int b = grow >> 11, s = grow & 2047;
        if (part == 1) {
          KhT[((size_t)(b * H_ + h) * HD_ + e) * S_ + s] = (f16)val;
        } else {
          f16* dst = (part == 0) ? Qh : Vh;
          dst[((size_t)(b * H_ + h) * S_ + s) * HD_ + e] = (f16)val;
        }
      }
}

// Output projection: out[8192][1024] = Ctxh * Woh + bo (f32 out).
__global__ __launch_bounds__(256) void gemm_out(
    const f16* __restrict__ Ctxh, const f16* __restrict__ Woh,
    const float* __restrict__ bo, float* __restrict__ out) {
  __shared__ f16 Alds[2 * 128 * LDA_S];
  __shared__ f16 Blds[2 * 32 * LDB_S];
  int tid = threadIdx.x, lane = tid & 31, wave = tid >> 5;
  int row0 = blockIdx.y * 128, col0 = blockIdx.x * 128;
  float8 acc[2][4] = {{zero8(), zero8(), zero8(), zero8()},
                      {zero8(), zero8(), zero8(), zero8()}};
  gemm_mainloop(Ctxh, Woh, D_, D_, row0, col0, Alds, Blds, acc, tid);

  int n = lane & 15, mh = (lane >> 4) * 8;
  int wm = (wave >> 1) * 32, wn = (wave & 1) * 64;
#pragma unroll
  for (int sub = 0; sub < 2; ++sub)
#pragma unroll
    for (int j = 0; j < 4; ++j)
#pragma unroll
      for (int r = 0; r < 8; ++r) {
        int grow = row0 + wm + sub * 16 + r + mh;
        int gcol = col0 + wn + j * 16 + n;
        out[(size_t)grow * D_ + gcol] = acc[sub][j][r] + bo[gcol];
      }
}

// ---------------------------------------------------------------- attention
// grid (S/128, H, B). 8 waves/block; wave owns a 16-row query tile.
// K arrives pre-transposed ([e][s]) so both K and V tiles are straight async
// B128 copies; tiles are double-buffered.
#define LDS_T 72   // 144B stride, 16B aligned

static __device__ __forceinline__ void attn_stage(
    const f16* __restrict__ KpT, const f16* __restrict__ Vp, int kt,
    f16* Kl, f16* Vl, int tid) {
  int c0 = tid, c1 = tid + 256;
  // KtT tile: 64 e-rows x 64 keys = 512 8-elt chunks
  { int er = c0 >> 3, k8 = (c0 & 7) * 8;
    cp16(&Kl[er * LDS_T + k8], &KpT[(size_t)er * S_ + kt * 64 + k8]); }
  { int er = c1 >> 3, k8 = (c1 & 7) * 8;
    cp16(&Kl[er * LDS_T + k8], &KpT[(size_t)er * S_ + kt * 64 + k8]); }
  // V tile: 64 key-rows x 64 e = 512 8-elt chunks
  { int kr = c0 >> 3, e8 = (c0 & 7) * 8;
    cp16(&Vl[kr * LDS_T + e8], &Vp[(size_t)(kt * 64 + kr) * HD_ + e8]); }
  { int kr = c1 >> 3, e8 = (c1 & 7) * 8;
    cp16(&Vl[kr * LDS_T + e8], &Vp[(size_t)(kt * 64 + kr) * HD_ + e8]); }
}

__global__ __launch_bounds__(256) void attn_flash(
    const f16* __restrict__ Qh, const f16* __restrict__ KhT,
    const f16* __restrict__ Vh, f16* __restrict__ Ctxh) {
  __shared__ f16 KtT[2 * 64 * LDS_T];   // [buf][e][key]
  __shared__ f16 Vt[2 * 64 * LDS_T];    // [buf][key][e]
  __shared__ f16 Pl[8][16][LDS_T];      // per-wave P tile [m][key]

  int b = blockIdx.z, h = blockIdx.y, qt = blockIdx.x;
  int tid = threadIdx.x, lane = tid & 31, wave = tid >> 5;
  size_t bhOff = (size_t)(b * H_ + h) * S_ * HD_;
  const f16* Qp  = Qh + bhOff;
  const f16* KpT = KhT + bhOff;   // [e][s]
  const f16* Vp  = Vh + bhOff;
  int q0 = qt * 128 + wave * 16;

  half16 qf0 = load_a_frag(Qp + (size_t)q0 * HD_, HD_, lane);
  half16 qf1 = load_a_frag(Qp + (size_t)q0 * HD_ + 32, HD_, lane);

  float8 o[4] = {zero8(), zero8(), zero8(), zero8()};
  float m_i[8], l_i[8];
#pragma unroll
  for (int r = 0; r < 8; ++r) { m_i[r] = -1e30f; l_i[r] = 0.f; }

  const int nTiles = S_ / 64;
  attn_stage(KpT, Vp, 0, KtT, Vt, tid);
  for (int kt = 0; kt < nTiles; ++kt) {
    f16* Kc = KtT + (kt & 1) * (64 * LDS_T);
    f16* Vc = Vt + (kt & 1) * (64 * LDS_T);
    if (kt + 1 < nTiles) {
      attn_stage(KpT, Vp, kt + 1,
                 KtT + ((kt + 1) & 1) * (64 * LDS_T),
                 Vt + ((kt + 1) & 1) * (64 * LDS_T), tid);
      WAIT_ASYNC(4);
    } else {
      WAIT_ASYNC(0);
    }
    __syncthreads();

    // scores: 16 x 64 = 4 N-tiles, K-dim = HD = 2 chained WMMAs each
    float8 s[4];
#pragma unroll
    for (int j = 0; j < 4; ++j) {
      half16 bk0 = load_b_frag(&Kc[j * 16], LDS_T, lane);
      float8 z = wmma_f16(qf0, bk0, zero8());
      half16 bk1 = load_b_frag(&Kc[32 * LDS_T + j * 16], LDS_T, lane);
      s[j] = wmma_f16(qf1, bk1, z);
    }

    // online softmax: row stats replicated across each 16-lane half
    float alpha[8];
#pragma unroll
    for (int r = 0; r < 8; ++r) {
      float mx = fmaxf(fmaxf(s[0][r], s[1][r]), fmaxf(s[2][r], s[3][r]));
      mx = fmaxf(mx, __shfl_xor(mx, 1));
      mx = fmaxf(mx, __shfl_xor(mx, 2));
      mx = fmaxf(mx, __shfl_xor(mx, 4));
      mx = fmaxf(mx, __shfl_xor(mx, 8));
      float mnew = fmaxf(m_i[r], mx);
      alpha[r] = __expf(m_i[r] - mnew);
      m_i[r] = mnew;
      float rs = 0.f;
#pragma unroll
      for (int j = 0; j < 4; ++j) {
        float p = __expf(s[j][r] - mnew);
        s[j][r] = p;
        rs += p;
      }
      rs += __shfl_xor(rs, 1);
      rs += __shfl_xor(rs, 2);
      rs += __shfl_xor(rs, 4);
      rs += __shfl_xor(rs, 8);
      l_i[r] = l_i[r] * alpha[r] + rs;
    }

    // P: C-layout -> A-layout via per-wave LDS round trip (DS ops are
    // in-order within a wave; asm barrier stops compiler reordering).
#pragma unroll
    for (int j = 0; j < 4; ++j)
#pragma unroll
      for (int r = 0; r < 8; ++r)
        Pl[wave][r + ((lane >> 4) << 3)][j * 16 + (lane & 15)] = (f16)s[j][r];
    asm volatile("" ::: "memory");

#pragma unroll
    for (int j = 0; j < 4; ++j)
#pragma unroll
      for (int r = 0; r < 8; ++r) o[j][r] *= alpha[r];

    half16 p0 = load_a_frag(&Pl[wave][0][0], LDS_T, lane);
    half16 p1 = load_a_frag(&Pl[wave][0][32], LDS_T, lane);
#pragma unroll
    for (int j = 0; j < 4; ++j) {
      half16 bv0 = load_b_frag(&Vc[j * 16], LDS_T, lane);
      o[j] = wmma_f16(p0, bv0, o[j]);
      half16 bv1 = load_b_frag(&Vc[32 * LDS_T + j * 16], LDS_T, lane);
      o[j] = wmma_f16(p1, bv1, o[j]);
    }
    __syncthreads();
  }

  // normalize and write ctx in [b][s][h*64+e] layout (f16) for final GEMM
#pragma unroll
  for (int j = 0; j < 4; ++j)
#pragma unroll
    for (int r = 0; r < 8; ++r) {
      int m = r + ((lane >> 4) << 3);
      float val = o[j][r] / l_i[r];
      int e = h * HD_ + j * 16 + (lane & 15);
      Ctxh[(size_t)(b * S_ + q0 + m) * D_ + e] = (f16)val;
    }
}

// ---------------------------------------------------------------- launcher
extern "C" void kernel_launch(void* const* d_in, const int* in_sizes, int n_in,
                              void* d_out, int out_size, void* d_ws, size_t ws_size,
                              hipStream_t stream) {
  const float* x  = (const float*)d_in[0];
  const float* Wq = (const float*)d_in[1];
  const float* bq = (const float*)d_in[2];
  const float* Wk = (const float*)d_in[3];
  const float* bk = (const float*)d_in[4];
  const float* Wv = (const float*)d_in[5];
  const float* bv = (const float*)d_in[6];
  const float* Wo = (const float*)d_in[7];
  const float* bo = (const float*)d_in[8];
  float* out = (float*)d_out;

  char* ws = (char*)d_ws;
  const size_t szX   = (size_t)B_ * S_ * D_ * sizeof(f16);       // 16 MiB
  const size_t szWq  = (size_t)D_ * 3 * D_ * sizeof(f16);        //  6 MiB
  const size_t szWo  = (size_t)D_ * D_ * sizeof(f16);            //  2 MiB
  const size_t szQKV = (size_t)B_ * H_ * S_ * HD_ * sizeof(f16); // 16 MiB each
  f16* Xh   = (f16*)(ws);
  f16* Wh   = (f16*)(ws + szX);
  f16* Woh  = (f16*)(ws + szX + szWq);
  f16* Qh   = (f16*)(ws + szX + szWq + szWo);
  f16* KhT  = (f16*)(ws + szX + szWq + szWo + szQKV);
  f16* Vh   = (f16*)(ws + szX + szWq + szWo + 2 * szQKV);
  f16* Ctxh = (f16*)(ws + szX + szWq + szWo + 3 * szQKV);

  {
    int n = B_ * S_ * D_;
    cvt_f32_to_f16<<<(n + 255) / 256, 256, 0, stream>>>(x, Xh, n);
  }
  {
    int n = D_ * 3 * D_;
    fuse_wqkv<<<(n + 255) / 256, 256, 0, stream>>>(Wq, Wk, Wv, Wh);
  }
  {
    int n = D_ * D_;
    cvt_f32_to_f16<<<(n + 255) / 256, 256, 0, stream>>>(Wo, Woh, n);
  }

  // QKV projection GEMM: (8192 x 1024) x (1024 x 3072)
  gemm_qkv<<<dim3(3 * D_ / 128, B_ * S_ / 128), 256, 0, stream>>>(
      Xh, Wh, bq, bk, bv, Qh, KhT, Vh);

  // flash attention
  attn_flash<<<dim3(S_ / 128, H_, B_), 256, 0, stream>>>(Qh, KhT, Vh, Ctxh);

  // output projection GEMM: (8192 x 1024) x (1024 x 1024) + bo
  gemm_out<<<dim3(D_ / 128, B_ * S_ / 128), 256, 0, stream>>>(Ctxh, Woh, bo, out);
}
